// TransformerNet_56856777064815
// MI455X (gfx1250) — matrix operand order repaired and checked by reading, not compiled, not run on hardware
//
#include <hip/hip_runtime.h>
#include <hip/hip_bf16.h>

#define NN 100000
#define NE 1600000
#define D1 256

typedef __attribute__((ext_vector_type(16))) _Float16 v16h;
typedef __attribute__((ext_vector_type(8)))  float    v8f;

union AF { v16h v; _Float16 e[16]; };
union CF { v8f  v; float    e[8];  };

// ---- order-preserving float<->uint encoding for atomic segment-max ----
__device__ __forceinline__ unsigned encf(float f) {
    unsigned u = __float_as_uint(f);
    return (u & 0x80000000u) ? ~u : (u | 0x80000000u);
}
__device__ __forceinline__ float decf(unsigned u) {
    return __uint_as_float((u & 0x80000000u) ? (u & 0x7fffffffu) : ~u);
}

__device__ __forceinline__ void zero_c(CF& c) {
#pragma unroll
    for (int r = 0; r < 8; ++r) c.e[r] = 0.f;
}

// =====================================================================
// K0: init per-node reduction state
// =====================================================================
__global__ void k_init(unsigned* mEnc1, float* s1den, unsigned* mEnc2, float* s2den) {
    int n = blockIdx.x * 256 + threadIdx.x;
    if (n >= NN) return;
    mEnc1[n] = 0u;   // encodes "most negative"
    s1den[n] = 0.f;
    mEnc2[n] = 0u;
    s2den[n] = 0.f;
}

// =====================================================================
// K1: layer-1 projections via WMMA.
// One wave per 16-node tile. A = x-tile (16x32 f16, only K=0,1 nonzero).
// For each weight matrix, 16 WMMA calls cover its 256 output channels.
// =====================================================================
__device__ __forceinline__ void proj1_quarter(
    const v16h a, const float* __restrict__ W, const float* __restrict__ bias,
    float* __restrict__ out, int tile, int m, int g) {
    for (int cb = 0; cb < 16; ++cb) {
        int col = cb * 16 + m;                 // output channel 0..255
        AF b;
#pragma unroll
        for (int t = 0; t < 16; ++t) b.e[t] = (_Float16)0.f;
        if (g == 0) {                          // B rows K=0,1 only (din=2)
            b.e[0] = (_Float16)W[col];         // W[0][col]
            b.e[1] = (_Float16)W[D1 + col];    // W[1][col]
        }
        CF c; zero_c(c);
        c.v = __builtin_amdgcn_wmma_f32_16x16x32_f16(
            false, a, false, b.v, (short)0, c.v, false, false);
        float bb = bias[col];
#pragma unroll
        for (int r = 0; r < 8; ++r) {
            int node = tile * 16 + r + 8 * g;  // C layout: M = r + 8*(lane/16)
            out[node * D1 + col] = c.e[r] + bb;
        }
    }
}

__global__ void k_proj1(const float* __restrict__ x,
                        const float* Wq, const float* bq,
                        const float* Wk, const float* bk,
                        const float* Wv, const float* bv_,
                        const float* Wsk, const float* bsk,
                        float* q1, float* k1, float* v1, float* h) {
    int tile = blockIdx.x;
    int lane = threadIdx.x;
    int m = lane & 15, g = lane >> 4;
    int node = tile * 16 + m;
    AF a;
#pragma unroll
    for (int t = 0; t < 16; ++t) a.e[t] = (_Float16)0.f;
    if (g == 0) {                              // A 16x32: lane g=0, t=0,1 -> K=0,1
        a.e[0] = (_Float16)x[node * 2 + 0];
        a.e[1] = (_Float16)x[node * 2 + 1];
    }
    proj1_quarter(a.v, Wq,  bq,  q1, tile, m, g);
    proj1_quarter(a.v, Wk,  bk,  k1, tile, m, g);
    proj1_quarter(a.v, Wv,  bv_, v1, tile, m, g);
    proj1_quarter(a.v, Wsk, bsk, h,  tile, m, g);   // skip path -> h accumulator
}

// =====================================================================
// K2: layer-1 edge logits. One wave32 per edge: 256-d dot via float4
// loads (8 f32/lane) + shuffle reduction; atomic segment-max on dst.
// =====================================================================
__global__ void k_logits1(const int* __restrict__ ei,
                          const float* __restrict__ q1, const float* __restrict__ k1,
                          float* __restrict__ alpha, unsigned* __restrict__ mEnc) {
    int wave = threadIdx.x >> 5, lane = threadIdx.x & 31;
    int e = blockIdx.x * 8 + wave;
    if (e >= NE) return;
    int s = ei[e], d = ei[NE + e];
    const float4* qv = (const float4*)(q1 + d * D1) + lane * 2;
    const float4* kv = (const float4*)(k1 + s * D1) + lane * 2;
    float4 qa = qv[0], qb = qv[1], ka = kv[0], kb = kv[1];
    float acc = qa.x * ka.x + qa.y * ka.y + qa.z * ka.z + qa.w * ka.w
              + qb.x * kb.x + qb.y * kb.y + qb.z * kb.z + qb.w * kb.w;
#pragma unroll
    for (int off = 16; off >= 1; off >>= 1) acc += __shfl_xor(acc, off, 32);
    if (lane == 0) {
        float al = acc * 0.0625f;              // 1/sqrt(256)
        alpha[e] = al;
        atomicMax(&mEnc[d], encf(al));
    }
}

// =====================================================================
// K3: e = exp(alpha - m[dst]); segment-sum into s1den; alpha <- e
// =====================================================================
__global__ void k_exp1(const int* __restrict__ ei, float* __restrict__ alpha,
                       const unsigned* __restrict__ mEnc, float* __restrict__ sden) {
    int e = blockIdx.x * 256 + threadIdx.x;
    if (e >= NE) return;
    int d = ei[NE + e];
    float ex = __expf(alpha[e] - decf(mEnc[d]));
    alpha[e] = ex;
    atomicAdd(&sden[d], ex);
}

// =====================================================================
// K4: h[dst] += (e/s[dst]) * v1[src]  — one wave per edge, 8 f32/lane
// =====================================================================
__global__ void k_agg1(const int* __restrict__ ei, const float* __restrict__ alpha,
                       const float* __restrict__ sden, const float* __restrict__ v1,
                       float* __restrict__ h) {
    int wave = threadIdx.x >> 5, lane = threadIdx.x & 31;
    int e = blockIdx.x * 8 + wave;
    if (e >= NE) return;
    int s = ei[e], d = ei[NE + e];
    float a = alpha[e] / sden[d];
    const float* vv = v1 + s * D1 + lane * 8;
    float* hh = h + d * D1 + lane * 8;
#pragma unroll
    for (int i = 0; i < 8; ++i) atomicAdd(&hh[i], a * vv[i]);
}

// =====================================================================
// K5: layer-2 projections via one WMMA chain. ReLU fused into A-load.
// B packs [Wq2|Wk2|Wv2|Ws2] (256x8) into 16 columns; K=256 in 8 steps.
// =====================================================================
__global__ void k_proj2(const float* __restrict__ h,
                        const float* Wq, const float* bq,
                        const float* Wk, const float* bk,
                        const float* Wv, const float* bv_,
                        const float* Wsk, const float* bsk,
                        float* q2, float* k2, float* v2, float* acc) {
    int tile = blockIdx.x;
    int lane = threadIdx.x;
    int m = lane & 15, g = lane >> 4;
    int node = tile * 16 + m;

    const float* Wsel = nullptr; const float* bsel = nullptr; float* osel = nullptr;
    int j = m & 1;
    if      (m < 2) { Wsel = Wq;  bsel = bq;  osel = q2;  }
    else if (m < 4) { Wsel = Wk;  bsel = bk;  osel = k2;  }
    else if (m < 6) { Wsel = Wv;  bsel = bv_; osel = v2;  }
    else if (m < 8) { Wsel = Wsk; bsel = bsk; osel = acc; }

    CF c; zero_c(c);
    for (int kb = 0; kb < 8; ++kb) {
        AF a, b;
#pragma unroll
        for (int t = 0; t < 16; ++t) {
            int K = kb * 32 + t + 16 * g;                    // A 16x32 lane map
            a.e[t] = (_Float16)fmaxf(h[node * D1 + K], 0.f); // fused ReLU
        }
#pragma unroll
        for (int t = 0; t < 16; ++t) b.e[t] = (_Float16)0.f;
        if (Wsel) {
#pragma unroll
            for (int t = 0; t < 16; ++t) {
                int K = kb * 32 + t + 16 * g;                // B 32x16 lane map
                b.e[t] = (_Float16)Wsel[K * 2 + j];
            }
        }
        c.v = __builtin_amdgcn_wmma_f32_16x16x32_f16(
            false, a.v, false, b.v, (short)0, c.v, false, false);
    }
    if (Wsel) {
        float bb = bsel[j];
#pragma unroll
        for (int r = 0; r < 8; ++r) {
            int nd = tile * 16 + r + 8 * g;
            osel[nd * 2 + j] = c.e[r] + bb;
        }
    }
}

// =====================================================================
// K6/K7/K8: layer-2 edge phase, d=2 — thread per edge
// =====================================================================
__global__ void k_logits2(const int* __restrict__ ei,
                          const float* __restrict__ q2, const float* __restrict__ k2,
                          float* __restrict__ alpha, unsigned* __restrict__ mEnc) {
    int e = blockIdx.x * 256 + threadIdx.x;
    if (e >= NE) return;
    int s = ei[e], d = ei[NE + e];
    float al = (q2[d * 2] * k2[s * 2] + q2[d * 2 + 1] * k2[s * 2 + 1]) * 0.70710678f;
    alpha[e] = al;
    atomicMax(&mEnc[d], encf(al));
}

__global__ void k_exp2(const int* __restrict__ ei, float* __restrict__ alpha,
                       const unsigned* __restrict__ mEnc, float* __restrict__ sden) {
    int e = blockIdx.x * 256 + threadIdx.x;
    if (e >= NE) return;
    int d = ei[NE + e];
    float ex = __expf(alpha[e] - decf(mEnc[d]));
    alpha[e] = ex;
    atomicAdd(&sden[d], ex);
}

__global__ void k_agg2(const int* __restrict__ ei, const float* __restrict__ alpha,
                       const float* __restrict__ sden, const float* __restrict__ v2,
                       float* __restrict__ acc) {
    int e = blockIdx.x * 256 + threadIdx.x;
    if (e >= NE) return;
    int s = ei[e], d = ei[NE + e];
    float a = alpha[e] / sden[d];
    atomicAdd(&acc[d * 2 + 0], a * v2[s * 2 + 0]);
    atomicAdd(&acc[d * 2 + 1], a * v2[s * 2 + 1]);
}

// =====================================================================
// K9: log_softmax over the 2 classes
// =====================================================================
__global__ void k_lsm(const float* __restrict__ acc, float* __restrict__ out) {
    int n = blockIdx.x * 256 + threadIdx.x;
    if (n >= NN) return;
    float a0 = acc[2 * n], a1 = acc[2 * n + 1];
    float mx = fmaxf(a0, a1);
    float l = mx + __logf(__expf(a0 - mx) + __expf(a1 - mx));
    out[2 * n + 0] = a0 - l;
    out[2 * n + 1] = a1 - l;
}

// =====================================================================
extern "C" void kernel_launch(void* const* d_in, const int* in_sizes, int n_in,
                              void* d_out, int out_size, void* d_ws, size_t ws_size,
                              hipStream_t stream) {
    const float* x   = (const float*)d_in[0];
    const int*   ei  = (const int*)  d_in[1];
    const float* Wq1 = (const float*)d_in[2];  const float* bq1 = (const float*)d_in[3];
    const float* Wk1 = (const float*)d_in[4];  const float* bk1 = (const float*)d_in[5];
    const float* Wv1 = (const float*)d_in[6];  const float* bv1 = (const float*)d_in[7];
    const float* Ws1 = (const float*)d_in[8];  const float* bs1 = (const float*)d_in[9];
    const float* Wq2 = (const float*)d_in[10]; const float* bq2 = (const float*)d_in[11];
    const float* Wk2 = (const float*)d_in[12]; const float* bk2 = (const float*)d_in[13];
    const float* Wv2 = (const float*)d_in[14]; const float* bv2 = (const float*)d_in[15];
    const float* Ws2 = (const float*)d_in[16]; const float* bs2 = (const float*)d_in[17];
    float* out = (float*)d_out;

    // ---- carve workspace ----
    char* p = (char*)d_ws;
    size_t off = 0;
    auto carve = [&](size_t bytes) {
        void* r = p + off;
        off += (bytes + 255) & ~(size_t)255;
        return r;
    };
    float*    q1     = (float*)   carve((size_t)NN * D1 * 4);
    float*    k1     = (float*)   carve((size_t)NN * D1 * 4);
    float*    v1     = (float*)   carve((size_t)NN * D1 * 4);
    float*    h      = (float*)   carve((size_t)NN * D1 * 4);
    float*    alpha1 = (float*)   carve((size_t)NE * 4);
    unsigned* mEnc1  = (unsigned*)carve((size_t)NN * 4);
    float*    s1den  = (float*)   carve((size_t)NN * 4);
    float*    q2     = (float*)   carve((size_t)NN * 2 * 4);
    float*    k2     = (float*)   carve((size_t)NN * 2 * 4);
    float*    v2     = (float*)   carve((size_t)NN * 2 * 4);
    float*    acc    = (float*)   carve((size_t)NN * 2 * 4);
    float*    alpha2 = (float*)   carve((size_t)NE * 4);
    unsigned* mEnc2  = (unsigned*)carve((size_t)NN * 4);
    float*    s2den  = (float*)   carve((size_t)NN * 4);

    const int nodeBlk = (NN + 255) / 256;        // 391
    const int tiles   = NN / 16;                 // 6250 (exact)
    const int edgeBlkW = NE / 8;                 // 200000 (wave per edge, 8/block)
    const int edgeBlkT = NE / 256;               // 6250 (thread per edge)

    k_init  <<<nodeBlk, 256, 0, stream>>>(mEnc1, s1den, mEnc2, s2den);
    k_proj1 <<<tiles, 32, 0, stream>>>(x, Wq1, bq1, Wk1, bk1, Wv1, bv1, Ws1, bs1,
                                       q1, k1, v1, h);
    k_logits1<<<edgeBlkW, 256, 0, stream>>>(ei, q1, k1, alpha1, mEnc1);
    k_exp1  <<<edgeBlkT, 256, 0, stream>>>(ei, alpha1, mEnc1, s1den);
    k_agg1  <<<edgeBlkW, 256, 0, stream>>>(ei, alpha1, s1den, v1, h);
    k_proj2 <<<tiles, 32, 0, stream>>>(h, Wq2, bq2, Wk2, bk2, Wv2, bv2, Ws2, bs2,
                                       q2, k2, v2, acc);
    k_logits2<<<edgeBlkT, 256, 0, stream>>>(ei, q2, k2, alpha2, mEnc2);
    k_exp2  <<<edgeBlkT, 256, 0, stream>>>(ei, alpha2, mEnc2, s2den);
    k_agg2  <<<edgeBlkT, 256, 0, stream>>>(ei, alpha2, s2den, v2, acc);
    k_lsm   <<<nodeBlk, 256, 0, stream>>>(acc, out);
}
